// PatchRouter_39754217292096
// MI455X (gfx1250) — compile-verified, matrix operand order checked by reading, not run on hardware
//
#include <hip/hip_runtime.h>

typedef _Float16 v16h __attribute__((ext_vector_type(16)));
typedef _Float16 v8h  __attribute__((ext_vector_type(8)));
typedef float    v8f  __attribute__((ext_vector_type(8)));
typedef float    v4f  __attribute__((ext_vector_type(4)));
typedef unsigned v4u  __attribute__((ext_vector_type(4)));
typedef unsigned v8u  __attribute__((ext_vector_type(8)));

#define NB    4        // batch
#define NC    128      // channels
#define NH    256
#define NW    256
#define NPIX  65536    // NH*NW
#define PH    128
#define PW    128
#define NQ    16384    // PH*PW
#define NE    64       // embed dim
#define NHEAD 8
#define NP    4
#define HD    8        // head dim

// ---------------------------------------------------------------------------
// WMMA fragment helpers.  A (16x32 f16) per documented CDNA5 layout:
//   lane l<16 : M=l,      halves 0..7 = K=kk..kk+7,   halves 8..15 = K=kk+16..kk+23
//   lane l>=16: M=l-16,   halves 0..7 = K=kk+8..kk+15,halves 8..15 = K=kk+24..kk+31
// B (32x16) addressed symmetrically from weights stored [N][K].
// ---------------------------------------------------------------------------
__device__ __forceinline__ v16h load_frag(const _Float16* p) {
    v8h lo = *(const v8h*)(p);
    v8h hi = *(const v8h*)(p + 16);
    v16h r;
#pragma unroll
    for (int i = 0; i < 8; ++i) { r[i] = lo[i]; r[8 + i] = hi[i]; }
    return r;
}

// ---------------------------------------------------------------------------
// Prep: fuse 1x1-conv weights with value projection, convert weights to f16.
// ---------------------------------------------------------------------------
__global__ __launch_bounds__(256) void prep_kernel(
    const float* __restrict__ w_v,  const float* __restrict__ satw,
    const float* __restrict__ satb, const float* __restrict__ b_v,
    const float* __restrict__ osmw, const float* __restrict__ osmb,
    const float* __restrict__ w_off,const float* __restrict__ b_off,
    const float* __restrict__ w_aw, const float* __restrict__ b_aw,
    const float* __restrict__ w_out,
    _Float16* __restrict__ WcS, _Float16* __restrict__ WcO,
    float* __restrict__ biasS, float* __restrict__ biasO,
    _Float16* __restrict__ Wqk, float* __restrict__ bqk,
    _Float16* __restrict__ Wout)
{
    const int tid = threadIdx.x;
    for (int idx = tid; idx < NE * NC; idx += 256) {      // Wc[e][c] = sum_k w_v[e][k]*conv_w[k][c]
        int e = idx >> 7, c = idx & 127;
        float s = 0.f, o = 0.f;
        for (int k = 0; k < NE; ++k) {
            float wv = w_v[e * NE + k];
            s += wv * satw[k * NC + c];
            o += wv * osmw[k * NC + c];
        }
        WcS[idx] = (_Float16)s; WcO[idx] = (_Float16)o;
    }
    for (int e = tid; e < NE; e += 256) {
        float s = 0.f, o = 0.f;
        for (int k = 0; k < NE; ++k) {
            float wv = w_v[e * NE + k];
            s += wv * satb[k];
            o += wv * osmb[k];
        }
        biasS[e] = s + b_v[e]; biasO[e] = o + b_v[e];
    }
    for (int idx = tid; idx < 96 * NE; idx += 256) {      // [w_off ; w_aw] rows
        int n = idx >> 6, k = idx & 63;
        float v = (n < 64) ? w_off[n * NE + k] : w_aw[(n - 64) * NE + k];
        Wqk[idx] = (_Float16)v;
    }
    for (int n = tid; n < 96; n += 256) bqk[n] = (n < 64) ? b_off[n] : b_aw[n - 64];
    for (int idx = tid; idx < NE * NE; idx += 256) Wout[idx] = (_Float16)w_out[idx];
}

// ---------------------------------------------------------------------------
// Score 1x1 conv (K=20) + LoFTR sinusoidal PE -> sc (B,E,PH*PW), q (B*NQ,E)
// ---------------------------------------------------------------------------
__global__ __launch_bounds__(256) void score_query_kernel(
    const float* __restrict__ score, const float* __restrict__ w,
    const float* __restrict__ bias,
    float* __restrict__ sc, float* __restrict__ q)
{
    int gid = blockIdx.x * 256 + threadIdx.x;             // < NB*NQ
    int b = gid >> 14, pix = gid & (NQ - 1);
    int ph = pix >> 7, pw = pix & 127;
    const float* sp = score + ((size_t)b * 20) * NQ + pix;
    float sv[20];
#pragma unroll
    for (int k = 0; k < 20; ++k) sv[k] = sp[(size_t)k * NQ];
    float xr = (float)(pw + 1), yr = (float)(ph + 1);
    for (int e = 0; e < NE; ++e) {
        float acc = bias[e];
#pragma unroll
        for (int k = 0; k < 20; ++k) acc += sv[k] * w[e * 20 + k];
        int m = e >> 2, r = e & 3;
        float div = expf(-(float)m * 0.57564627324851148f);  // ln(1e4)/16
        float ang = ((r < 2) ? xr : yr) * div;
        float pe  = (r & 1) ? cosf(ang) : sinf(ang);
        sc[((size_t)(b * NE + e)) * NQ + pix] = acc;
        q[(size_t)gid * NE + e] = acc + pe;
    }
}

// ---------------------------------------------------------------------------
// WMMA GEMM fed by the Tensor Data Mover:
//   vp(B,NPIX,64) = f(B,128,NPIX)^T @ Wc(64,128)^T + bias  (f16 out)
// Block = 8 waves, 64 pixel-rows.  TDM DMAs the (64 pix x 64 chan) f32 tile
// into an LDS bounce buffer (2 chunks over K=128); threads convert to the
// f16 fragment layout; waves 2-way split the N=64 output columns.
// ---------------------------------------------------------------------------
#define VAS 136   // padded LDS row stride (halves), 272B = 16B aligned, bank-staggered
__global__ __launch_bounds__(256) void vproj_gemm_kernel(
    const float* __restrict__ f, const _Float16* __restrict__ Wc,
    const float* __restrict__ bias, _Float16* __restrict__ vp)
{
    __shared__ float    fstage[64 * 64];   // 16 KB TDM landing pad (f32 tile chunk)
    __shared__ _Float16 aT[64 * VAS];      // 17 KB A fragments (f16)
    __shared__ _Float16 wT[64 * VAS];      // 17 KB B fragments (f16)
    const int tid = threadIdx.x, blk = blockIdx.x;
    const int b = blk >> 10;                     // 1024 blocks per batch
    const int pixBase = (blk & 1023) << 6;
    const size_t fBase = ((size_t)b * NC) * NPIX;

    for (int idx = tid; idx < NE * NC; idx += 256) {   // stage weights (once)
        int e = idx >> 7, c = idx & 127;
        wT[e * VAS + c] = Wc[idx];
    }

    // K staged in two 64-channel chunks through the TDM
    const unsigned ldsAddr = (unsigned)(unsigned long long)(&fstage[0]); // flat->LDS: low 32 bits
    for (int c2 = 0; c2 < 2; ++c2) {
        __syncthreads();                               // fstage free for reuse
        if (tid < 32) {                                // wave 0 drives the DMA
            unsigned long long ga = (unsigned long long)(const void*)
                (f + fBase + (size_t)(c2 * 64) * NPIX + pixBase);
            v4u g0;                                    // D# group 0 (ISA 8.3)
            g0[0] = 1u;                                // count=1, user mode
            g0[1] = ldsAddr;                           // lds_addr[31:0]
            g0[2] = (unsigned)(ga & 0xFFFFFFFFu);      // global_addr[31:0]
            g0[3] = (unsigned)((ga >> 32) & 0x01FFFFFFu) | (2u << 30); // ga[56:32] | type=2
            v8u g1;                                    // D# group 1 (ISA 8.4)
            g1[0] = (2u << 16);                        // wg_mask=0, data_size=4B
            g1[1] = 0u;                                // tensor_dim0[15:0]=0 (65536 lo), abar=0
            g1[2] = 1u | (64u << 16);                  // tensor_dim0 hi=1, tensor_dim1 lo=64
            g1[3] = (64u << 16);                       // tensor_dim1 hi=0, tile_dim0=64
            g1[4] = 64u;                               // tile_dim1=64, tile_dim2=0
            g1[5] = 65536u;                            // tensor_dim0_stride[31:0]
            g1[6] = 0u;                                // stride hi, dim1_stride lo
            g1[7] = 0u;
            asm volatile("tensor_load_to_lds %0, %1"
                         :: "s"(g0), "s"(g1) : "memory");
            __builtin_amdgcn_s_wait_tensorcnt(0);
        }
        __syncthreads();                               // publish fstage to all waves
        for (int idx = tid; idx < 64 * 64; idx += 256) {
            int c = idx >> 6, p = idx & 63;            // TDM wrote [chan][pix] contiguously
            aT[p * VAS + (c2 << 6) + c] = (_Float16)fstage[idx];
        }
    }
    __syncthreads();

    const int wv = tid >> 5, lane = tid & 31;
    const int lr = lane & 15, hi = lane >> 4;
    const int m0 = (wv & 3) << 4;                      // 4 M-tiles
    const int nbase = (wv >> 2) << 5;                  // 2 N-halves
    v8f z = {};
    v8f acc[2] = {z, z};
#pragma unroll
    for (int kk = 0; kk < NC; kk += 32) {
        v16h a = load_frag(&aT[(m0 + lr) * VAS + kk + hi * 8]);
#pragma unroll
        for (int t = 0; t < 2; ++t) {
            v16h bm = load_frag(&wT[(nbase + (t << 4) + lr) * VAS + kk + hi * 8]);
            acc[t] = __builtin_amdgcn_wmma_f32_16x16x32_f16(
                false, a, false, bm, (short)0, acc[t], false, false);
        }
    }
    const size_t outRow = (size_t)b * NPIX + pixBase;
#pragma unroll
    for (int t = 0; t < 2; ++t) {
        int n = nbase + (t << 4) + lr;
        float bn = bias[n];
#pragma unroll
        for (int r = 0; r < 8; ++r) {
            int m = m0 + r + (hi << 3);
            vp[(outRow + m) * NE + n] = (_Float16)(acc[t][r] + bn);
        }
    }
}

// ---------------------------------------------------------------------------
// WMMA GEMM: offaw(B*NQ,96) = q(B*NQ,64) @ Wqk(96,64)^T + bqk   (f32 out)
// ---------------------------------------------------------------------------
#define QAS 72
__global__ __launch_bounds__(256) void qproj_gemm_kernel(
    const float* __restrict__ q, const _Float16* __restrict__ Wqk,
    const float* __restrict__ bqk, float* __restrict__ offaw)
{
    __shared__ _Float16 aT[128 * QAS];
    __shared__ _Float16 wT[96 * QAS];
    const int tid = threadIdx.x;
    const int rowBase = blockIdx.x << 7;
    for (int idx = tid; idx < 128 * 64; idx += 256) {
        int r = idx >> 6, k = idx & 63;
        aT[r * QAS + k] = (_Float16)q[(size_t)(rowBase + r) * NE + k];
    }
    for (int idx = tid; idx < 96 * 64; idx += 256) {
        int n = idx >> 6, k = idx & 63;
        wT[n * QAS + k] = Wqk[idx];
    }
    __syncthreads();
    const int wv = tid >> 5, lane = tid & 31;
    const int lr = lane & 15, hi = lane >> 4;
    const int m0 = wv << 4;
    v8f z = {};
    v8f acc[6] = {z, z, z, z, z, z};
#pragma unroll
    for (int kk = 0; kk < 64; kk += 32) {
        v16h a = load_frag(&aT[(m0 + lr) * QAS + kk + hi * 8]);
#pragma unroll
        for (int t = 0; t < 6; ++t) {
            v16h bm = load_frag(&wT[((t << 4) + lr) * QAS + kk + hi * 8]);
            acc[t] = __builtin_amdgcn_wmma_f32_16x16x32_f16(
                false, a, false, bm, (short)0, acc[t], false, false);
        }
    }
#pragma unroll
    for (int t = 0; t < 6; ++t) {
        int n = (t << 4) + lr;
        float bn = bqk[n];
#pragma unroll
        for (int r = 0; r < 8; ++r) {
            int m = m0 + r + (hi << 3);
            offaw[(size_t)(rowBase + m) * 96 + n] = acc[t][r] + bn;
        }
    }
}

// ---------------------------------------------------------------------------
// Deformable sampling: same offsets/weights for both value tensors (reference
// passes the same q to both MSDA calls) -> o_sum = sum_p aw_p*(sat+osm).
// Thread per (b, query, head); taps are 16B-aligned v8h gathers.
// ---------------------------------------------------------------------------
__global__ __launch_bounds__(256) void msda_sample_kernel(
    const float* __restrict__ offaw, const _Float16* __restrict__ vpS,
    const _Float16* __restrict__ vpO, _Float16* __restrict__ osum)
{
    int gid = blockIdx.x * 256 + threadIdx.x;             // < NB*NQ*NHEAD
    int h = gid & 7;
    int qi = (gid >> 3) & (NQ - 1);
    int b  = gid >> 17;
    int row = (b << 14) + qi;
    const float* oa = offaw + (size_t)row * 96;
    float av[4];
#pragma unroll
    for (int p = 0; p < NP; ++p) av[p] = oa[64 + h * 4 + p];
    float mx = fmaxf(fmaxf(av[0], av[1]), fmaxf(av[2], av[3]));
    float s = 0.f;
#pragma unroll
    for (int p = 0; p < NP; ++p) { av[p] = expf(av[p] - mx); s += av[p]; }
    float inv = 1.f / s;
    float rx = (float)(qi & 255) * (1.0f / 255.0f);        // ref points (cg, rg)
    float ry = (float)(qi >> 8) * (4.0f / 255.0f);
    float acc[HD] = {0.f, 0.f, 0.f, 0.f, 0.f, 0.f, 0.f, 0.f};
    const size_t vb = ((size_t)b * NPIX) * NE + h * HD;
#pragma unroll
    for (int p = 0; p < NP; ++p) {
        float wp = av[p] * inv;
        float ox = oa[(h * 4 + p) * 2 + 0], oy = oa[(h * 4 + p) * 2 + 1];
        float x = (rx + ox * (1.f / 256.f)) * 256.f - 0.5f;
        float y = (ry + oy * (1.f / 256.f)) * 256.f - 0.5f;
        float x0f = floorf(x), y0f = floorf(y);
        float dx = x - x0f, dy = y - y0f;
        int x0 = (int)x0f, y0 = (int)y0f;
#pragma unroll
        for (int t = 0; t < 4; ++t) {
            int tx = t & 1, ty = t >> 1;
            int xi = x0 + tx, yi = y0 + ty;
            float wgt = (tx ? dx : 1.f - dx) * (ty ? dy : 1.f - dy) * wp;
            bool valid = (xi >= 0) && (xi < NW) && (yi >= 0) && (yi < NH);
            if (valid && wgt != 0.f) {
                size_t base = vb + (size_t)(yi * NW + xi) * NE;
                v8h sA = *(const v8h*)(vpS + base);
                v8h sB = *(const v8h*)(vpO + base);
#pragma unroll
                for (int d = 0; d < HD; ++d)
                    acc[d] += wgt * ((float)sA[d] + (float)sB[d]);
            }
        }
    }
    v8h out;
#pragma unroll
    for (int d = 0; d < HD; ++d) out[d] = (_Float16)acc[d];
    *(v8h*)(osum + (size_t)row * NE + h * HD) = out;
}

// ---------------------------------------------------------------------------
// WMMA GEMM + fused residual epilogue:
//   attended(B,E,NQ) = osum @ Wout^T + 2*b_out + 2*q + sc
// ---------------------------------------------------------------------------
__global__ __launch_bounds__(256) void outproj_gemm_kernel(
    const _Float16* __restrict__ osum, const _Float16* __restrict__ Wout,
    const float* __restrict__ b_out, const float* __restrict__ q,
    const float* __restrict__ sc, float* __restrict__ att)
{
    __shared__ _Float16 aT[128 * QAS];
    __shared__ _Float16 wT[64 * QAS];
    const int tid = threadIdx.x;
    const int rowBase = blockIdx.x << 7;
    for (int idx = tid; idx < 128 * 64; idx += 256) {
        int r = idx >> 6, k = idx & 63;
        aT[r * QAS + k] = osum[(size_t)(rowBase + r) * NE + k];
    }
    for (int idx = tid; idx < 64 * 64; idx += 256) {
        int n = idx >> 6, k = idx & 63;
        wT[n * QAS + k] = Wout[idx];
    }
    __syncthreads();
    const int wv = tid >> 5, lane = tid & 31;
    const int lr = lane & 15, hi = lane >> 4;
    const int m0 = wv << 4;
    v8f z = {};
    v8f acc[4] = {z, z, z, z};
#pragma unroll
    for (int kk = 0; kk < 64; kk += 32) {
        v16h a = load_frag(&aT[(m0 + lr) * QAS + kk + hi * 8]);
#pragma unroll
        for (int t = 0; t < 4; ++t) {
            v16h bm = load_frag(&wT[((t << 4) + lr) * QAS + kk + hi * 8]);
            acc[t] = __builtin_amdgcn_wmma_f32_16x16x32_f16(
                false, a, false, bm, (short)0, acc[t], false, false);
        }
    }
#pragma unroll
    for (int t = 0; t < 4; ++t) {
        int n = (t << 4) + lr;
        float bo2 = 2.f * b_out[n];
#pragma unroll
        for (int r = 0; r < 8; ++r) {
            int m = m0 + r + (hi << 3);
            int row = rowBase + m;
            int b = row >> 14, pix = row & (NQ - 1);
            size_t cidx = ((size_t)(b * NE + n)) * NQ + pix;
            att[cidx] = acc[t][r] + bo2 + 2.f * q[(size_t)row * NE + n] + sc[cidx];
        }
    }
}

// ---------------------------------------------------------------------------
// Downsampling convs (stride 2, pad 1) + router softmax
// ---------------------------------------------------------------------------
__global__ __launch_bounds__(256) void conv1_kernel(
    const float* __restrict__ att, const float* __restrict__ w,
    const float* __restrict__ bias, float* __restrict__ out)
{
    int gid = blockIdx.x * 256 + threadIdx.x;   // < 4*32*64*64
    int ow = gid & 63, oh = (gid >> 6) & 63, oc = (gid >> 12) & 31, b = gid >> 17;
    float acc = bias[oc];
    for (int ic = 0; ic < 64; ++ic) {
        const float* ip = att + ((size_t)(b * 64 + ic)) * NQ;
        const float* wp = w + (oc * 64 + ic) * 9;
#pragma unroll
        for (int kh = 0; kh < 3; ++kh) {
            int ih = oh * 2 + kh - 1;
            if (ih < 0 || ih >= 128) continue;
#pragma unroll
            for (int kw = 0; kw < 3; ++kw) {
                int iw = ow * 2 + kw - 1;
                if (iw < 0 || iw >= 128) continue;
                acc += ip[ih * 128 + iw] * wp[kh * 3 + kw];
            }
        }
    }
    out[gid] = acc;   // no relu on conv1 (matches reference)
}

__global__ __launch_bounds__(256) void conv2_kernel(
    const float* __restrict__ in, const float* __restrict__ w,
    const float* __restrict__ bias, float* __restrict__ out)
{
    int gid = blockIdx.x * 256 + threadIdx.x;   // < 4*16*32*32
    int ow = gid & 31, oh = (gid >> 5) & 31, oc = (gid >> 10) & 15, b = gid >> 14;
    float acc = bias[oc];
    for (int ic = 0; ic < 32; ++ic) {
        const float* ip = in + ((size_t)(b * 32 + ic)) * 4096;
        const float* wp = w + (oc * 32 + ic) * 9;
#pragma unroll
        for (int kh = 0; kh < 3; ++kh) {
            int ih = oh * 2 + kh - 1;
            if (ih < 0 || ih >= 64) continue;
#pragma unroll
            for (int kw = 0; kw < 3; ++kw) {
                int iw = ow * 2 + kw - 1;
                if (iw < 0 || iw >= 64) continue;
                acc += ip[ih * 64 + iw] * wp[kh * 3 + kw];
            }
        }
    }
    out[gid] = fmaxf(acc, 0.f);
}

__global__ __launch_bounds__(256) void conv3_softmax_kernel(
    const float* __restrict__ in, const float* __restrict__ w,
    const float* __restrict__ bias, float* __restrict__ wts)
{
    int gid = blockIdx.x * 256 + threadIdx.x;   // < 4*16*16
    if (gid >= NB * 256) return;
    int ow = gid & 15, oh = (gid >> 4) & 15, b = gid >> 8;
    float o[2] = {bias[0], bias[1]};
    for (int ic = 0; ic < 16; ++ic) {
        const float* ip = in + ((size_t)(b * 16 + ic)) * 1024;
#pragma unroll
        for (int kh = 0; kh < 3; ++kh) {
            int ih = oh * 2 + kh - 1;
            if (ih < 0 || ih >= 32) continue;
#pragma unroll
            for (int kw = 0; kw < 3; ++kw) {
                int iw = ow * 2 + kw - 1;
                if (iw < 0 || iw >= 32) continue;
                float v = ip[ih * 32 + iw];
                o[0] += v * w[(0 * 16 + ic) * 9 + kh * 3 + kw];
                o[1] += v * w[(1 * 16 + ic) * 9 + kh * 3 + kw];
            }
        }
    }
    o[0] = fmaxf(o[0], 0.f); o[1] = fmaxf(o[1], 0.f);
    const float scale = rsqrtf(32768.0f);       // 1/sqrt(D), D = C*PS*PS
    float l0 = o[0] * scale, l1 = o[1] * scale;
    float m = fmaxf(l0, l1);
    float e0 = expf(l0 - m), e1 = expf(l1 - m);
    float invs = 1.f / (e0 + e1);
    int n = oh * 16 + ow;
    wts[((size_t)(b * 256 + n)) * 2 + 0] = e0 * invs;
    wts[((size_t)(b * 256 + n)) * 2 + 1] = e1 * invs;
}

// ---------------------------------------------------------------------------
// Streaming router: out = w0*f1 + w1*f2 per 16x16 patch (b128 vector path)
// ---------------------------------------------------------------------------
__global__ __launch_bounds__(256) void route_kernel(
    const v4f* __restrict__ f1, const v4f* __restrict__ f2,
    const float* __restrict__ wts, v4f* __restrict__ out)
{
    size_t gid = (size_t)blockIdx.x * 256 + threadIdx.x;   // < NB*NC*NPIX/4
    size_t lin = gid << 2;
    int b = (int)(lin >> 23);
    int rem = (int)(lin & 8388607);
    int pixel = rem & (NPIX - 1);
    int h = pixel >> 8, ww = pixel & 255;
    int n = (h >> 4) * 16 + (ww >> 4);
    float w0 = wts[((size_t)(b * 256 + n)) * 2 + 0];
    float w1 = wts[((size_t)(b * 256 + n)) * 2 + 1];
    out[gid] = w0 * f1[gid] + w1 * f2[gid];
}

// ---------------------------------------------------------------------------
extern "C" void kernel_launch(void* const* d_in, const int* in_sizes, int n_in,
                              void* d_out, int out_size, void* d_ws, size_t ws_size,
                              hipStream_t stream) {
    (void)in_sizes; (void)n_in; (void)out_size; (void)ws_size;
    const float* f1     = (const float*)d_in[0];
    const float* f2     = (const float*)d_in[1];
    const float* score  = (const float*)d_in[2];
    const float* scw    = (const float*)d_in[3];
    const float* scb    = (const float*)d_in[4];
    const float* satw   = (const float*)d_in[5];
    const float* satb   = (const float*)d_in[6];
    const float* osmw   = (const float*)d_in[7];
    const float* osmb   = (const float*)d_in[8];
    const float* w_off  = (const float*)d_in[9];
    const float* b_off  = (const float*)d_in[10];
    const float* w_aw   = (const float*)d_in[11];
    const float* b_aw   = (const float*)d_in[12];
    const float* w_v    = (const float*)d_in[13];
    const float* b_v    = (const float*)d_in[14];
    const float* w_out  = (const float*)d_in[15];
    const float* b_out  = (const float*)d_in[16];
    const float* c1w    = (const float*)d_in[17];
    const float* c1b    = (const float*)d_in[18];
    const float* c2w    = (const float*)d_in[19];
    const float* c2b    = (const float*)d_in[20];
    const float* c3w    = (const float*)d_in[21];
    const float* c3b    = (const float*)d_in[22];

    char* ws = (char*)d_ws;
    size_t off = 0;
    auto carve = [&](size_t bytes) -> void* {
        void* p = ws + off;
        off = (off + bytes + 255) & ~(size_t)255;
        return p;
    };
    float*    sc    = (float*)   carve((size_t)NB * NE * NQ * 4);     // 16 MB
    float*    q     = (float*)   carve((size_t)NB * NQ * NE * 4);     // 16 MB
    float*    offaw = (float*)   carve((size_t)NB * NQ * 96 * 4);     // 24 MB
    _Float16* vpS   = (_Float16*)carve((size_t)NB * NPIX * NE * 2);   // 32 MB
    _Float16* vpO   = (_Float16*)carve((size_t)NB * NPIX * NE * 2);   // 32 MB
    _Float16* osum  = (_Float16*)carve((size_t)NB * NQ * NE * 2);     //  8 MB
    float*    att   = (float*)   carve((size_t)NB * NE * NQ * 4);     // 16 MB
    float*    c1    = (float*)   carve((size_t)NB * 32 * 4096 * 4);   //  2 MB
    float*    c2    = (float*)   carve((size_t)NB * 16 * 1024 * 4);
    _Float16* WcS   = (_Float16*)carve(NE * NC * 2);
    _Float16* WcO   = (_Float16*)carve(NE * NC * 2);
    _Float16* Wqk   = (_Float16*)carve(96 * NE * 2);
    _Float16* Wout  = (_Float16*)carve(NE * NE * 2);
    float*    biasS = (float*)   carve(NE * 4);
    float*    biasO = (float*)   carve(NE * 4);
    float*    bqk   = (float*)   carve(96 * 4);

    float* out_routed = (float*)d_out;                    // (B,C,H,W)
    float* out_w      = out_routed + (size_t)NB * NC * NPIX;  // (B,256,2)

    prep_kernel<<<1, 256, 0, stream>>>(w_v, satw, satb, b_v, osmw, osmb,
                                       w_off, b_off, w_aw, b_aw, w_out,
                                       WcS, WcO, biasS, biasO, Wqk, bqk, Wout);
    score_query_kernel<<<(NB * NQ) / 256, 256, 0, stream>>>(score, scw, scb, sc, q);
    vproj_gemm_kernel<<<(NB * NPIX) / 64, 256, 0, stream>>>(f1, WcS, biasS, vpS);
    vproj_gemm_kernel<<<(NB * NPIX) / 64, 256, 0, stream>>>(f2, WcO, biasO, vpO);
    qproj_gemm_kernel<<<(NB * NQ) / 128, 256, 0, stream>>>(q, Wqk, bqk, offaw);
    msda_sample_kernel<<<(NB * NQ * NHEAD) / 256, 256, 0, stream>>>(offaw, vpS, vpO, osum);
    outproj_gemm_kernel<<<(NB * NQ) / 128, 256, 0, stream>>>(osum, Wout, b_out, q, sc, att);
    conv1_kernel<<<(NB * 32 * 4096) / 256, 256, 0, stream>>>(att, c1w, c1b, c1);
    conv2_kernel<<<(NB * 16 * 1024) / 256, 256, 0, stream>>>(c1, c2w, c2b, c2);
    conv3_softmax_kernel<<<(NB * 256 + 255) / 256, 256, 0, stream>>>(c2, c3w, c3b, out_w);
    route_kernel<<<(NB * NC * NPIX / 4) / 256, 256, 0, stream>>>(
        (const v4f*)f1, (const v4f*)f2, out_w, (v4f*)out_routed);
}
